// AnchorDML_75737453298236
// MI455X (gfx1250) — compile-verified, matrix operand order checked by reading, not run on hardware
//
#include <hip/hip_runtime.h>
#include <hip/hip_bf16.h>
#include <math.h>

// ---------------------------------------------------------------------------
// AnchorDML on MI455X (gfx1250): all GEMMs via v_wmma_f32_16x16x32_bf16
//   xe = mish(mish(x@W1+b1)@W2+b2)      [8192,512]
//   se = mish(mish(s@W1+b1)@W2+b2)      [4096,512]
//   dist = sqrt(max(|xe|^2+|se|^2-2 xe.se^T,0))   [8192,4096] (bf16 scratch)
//   out = log_softmax(tanh(dist@Wp+bp)) [8192,100]
//
// Big GEMMs: 256-thread blocks, 64x128 block tile, K chunked at 128 through
// DOUBLE-BUFFERED LDS; staging uses GLOBAL_LOAD_ASYNC_TO_LDS_B128 (ASYNCcnt)
// overlapping the chunk-(k+1) DMA with the chunk-k WMMA burst.
// 32x32 per-wave register tile -> 4 WMMAs per K-step.
// ---------------------------------------------------------------------------

typedef __attribute__((ext_vector_type(16))) __bf16 v16bf;
typedef __attribute__((ext_vector_type(8)))  __bf16 v8bf;
typedef __attribute__((ext_vector_type(8)))  float  v8f;

// async-load builtin operand types: vector_size(16) int, addrspace-qualified
typedef int v4i __attribute__((vector_size(16)));
typedef __attribute__((address_space(1))) v4i AS1v4i;
typedef __attribute__((address_space(3))) v4i AS3v4i;

#define N_ROWS 8192
#define M_ROWS 4096
#define DDIM   512
#define CDIM   100
#define CPAD   128
#define KC     128   // K chunk staged in LDS for the big GEMMs

#if defined(__has_builtin)
#if __has_builtin(__builtin_amdgcn_global_load_async_to_lds_b128) && \
    __has_builtin(__builtin_amdgcn_s_wait_asynccnt)
#define USE_ASYNC_LDS 1
#endif
#endif
#ifndef USE_ASYNC_LDS
#define USE_ASYNC_LDS 0
#endif

// ---- WMMA helpers ---------------------------------------------------------

// 16-bit A/B fragment (16x32 / 32x16), per CDNA5 ISA §7.12.2:
//   lanes 0-15 : row r = lane,     K = kb+{0..7, 16..23}
//   lanes 16-31: row r = lane-16,  K = kb+{8..15, 24..31}
__device__ __forceinline__ v16bf frag_ld(const __bf16* base, int ld, int kb) {
  const int lane = threadIdx.x & 31;
  const int r    = lane & 15;
  const int k0   = kb + ((lane >> 4) << 3);
  const __bf16* p = base + (size_t)r * ld + k0;
  v8bf lo = *(const v8bf*)(p);
  v8bf hi = *(const v8bf*)(p + 16);
  v16bf f;
#pragma unroll
  for (int i = 0; i < 8; ++i) { f[i] = lo[i]; f[i + 8] = hi[i]; }
  return f;
}

__device__ __forceinline__ v8f wmma_bf16(v16bf a, v16bf b, v8f c) {
  return __builtin_amdgcn_wmma_f32_16x16x32_bf16(
      /*neg_a=*/false, a, /*neg_b=*/false, b,
      /*c_mod=*/(short)0, c, /*reuse_a=*/false, /*reuse_b=*/false);
}

__device__ __forceinline__ float mishf(float x) {
  float sp = (x > 20.0f) ? x : log1pf(__expf(x));
  return x * tanhf(sp);
}

// ---- staging: 16B per thread-iteration, async DMA to LDS when available ---

__device__ __forceinline__ void stage_chunk(const __bf16* __restrict__ g, int ld,
                                            int k0, __bf16* lds, int rows) {
  const int tid = threadIdx.x;
#pragma unroll
  for (int i = tid; i < rows * (KC / 8); i += 256) {
    int r = i >> 4, c = i & 15;                 // KC/8 = 16 float4 per row
    const __bf16* src = g + (size_t)r * ld + k0 + c * 8;
    __bf16* dst = lds + (size_t)r * KC + c * 8;
#if USE_ASYNC_LDS
    __builtin_amdgcn_global_load_async_to_lds_b128(
        (AS1v4i*)(void*)src, (AS3v4i*)(void*)dst, /*offset=*/0, /*cpol=*/0);
#else
    *(float4*)dst = *(const float4*)src;
#endif
  }
}

__device__ __forceinline__ void stage_fence() {
#if USE_ASYNC_LDS
  __builtin_amdgcn_s_wait_asynccnt(0);
#endif
  __syncthreads();
}

// ---- tiled GEMM core: block 64x128, wave 32x32, double-buffered K chunks --
// Aglob: rows [rowbase..+64) of [*, lda] bf16 row-major
// Bglob: rows [colbase..+128) of [*, ldb] bf16 row-major (i.e. B^T rows)
// As: 2 x 64*KC, Bs: 2 x 128*KC (LDS)
__device__ __forceinline__ void gemm64x128(const __bf16* __restrict__ Aglob,
                                           const __bf16* __restrict__ Bglob,
                                           int lda, int ldb, int ktot,
                                           __bf16* As, __bf16* Bs,
                                           v8f acc[2][2]) {
  const int rw = (threadIdx.x >> 5) >> 2;   // 0..1 row-wave
  const int cw = (threadIdx.x >> 5) & 3;    // 0..3 col-wave
  const int nchunk = ktot / KC;

  stage_chunk(Aglob, lda, 0, As, 64);
  stage_chunk(Bglob, ldb, 0, Bs, 128);
  stage_fence();

  for (int c = 0; c < nchunk; ++c) {
    const int cur = c & 1;
    const int nxt = cur ^ 1;
    if (c + 1 < nchunk) {                      // prefetch chunk c+1 via DMA
      stage_chunk(Aglob, lda, (c + 1) * KC, As + (size_t)nxt * 64 * KC, 64);
      stage_chunk(Bglob, ldb, (c + 1) * KC, Bs + (size_t)nxt * 128 * KC, 128);
    }
    const __bf16* Ab = As + (size_t)cur * 64 * KC;
    const __bf16* Bb = Bs + (size_t)cur * 128 * KC;
#pragma unroll
    for (int kk = 0; kk < KC; kk += 32) {
      v16bf a0 = frag_ld(Ab + (size_t)(rw * 32 + 0)  * KC, KC, kk);
      v16bf a1 = frag_ld(Ab + (size_t)(rw * 32 + 16) * KC, KC, kk);
      v16bf b0 = frag_ld(Bb + (size_t)(cw * 32 + 0)  * KC, KC, kk);
      v16bf b1 = frag_ld(Bb + (size_t)(cw * 32 + 16) * KC, KC, kk);
      acc[0][0] = wmma_bf16(a0, b0, acc[0][0]);
      acc[0][1] = wmma_bf16(a0, b1, acc[0][1]);
      acc[1][0] = wmma_bf16(a1, b0, acc[1][0]);
      acc[1][1] = wmma_bf16(a1, b1, acc[1][1]);
    }
    stage_fence();
  }
}

// ---- conversion / transpose kernels --------------------------------------

__global__ __launch_bounds__(256) void cvt_bf16(const float* __restrict__ src,
                                                __bf16* __restrict__ dst, int n) {
  int i = (blockIdx.x * 256 + threadIdx.x) * 4;
  if (i < n) {
    float4 v = *(const float4*)(src + i);
    dst[i + 0] = (__bf16)v.x;
    dst[i + 1] = (__bf16)v.y;
    dst[i + 2] = (__bf16)v.z;
    dst[i + 3] = (__bf16)v.w;
  }
}

// src: f32 [K,N] row-major;  dst: bf16 [Npad,K], zero-padded rows n>=N
__global__ __launch_bounds__(256) void transpose_cvt(const float* __restrict__ src,
                                                     __bf16* __restrict__ dst,
                                                     int K, int N, int Npad) {
  int idx = blockIdx.x * 256 + threadIdx.x;
  if (idx >= Npad * K) return;
  int n = idx / K;
  int k = idx - n * K;
  float v = (n < N) ? src[(size_t)k * N + n] : 0.0f;
  dst[idx] = (__bf16)v;
}

// ---- encode layer: OUT = mish(A @ W + b) ---------------------------------
// grid = (R/64, 512/128), block = 256

__global__ __launch_bounds__(256)
void gemm_bias_mish(const __bf16* __restrict__ A, const __bf16* __restrict__ Wt,
                    const float* __restrict__ bias, __bf16* __restrict__ OUT) {
  __shared__ __align__(16) __bf16 As[2 * 64 * KC];
  __shared__ __align__(16) __bf16 Bs[2 * 128 * KC];
  const int tid     = threadIdx.x;
  const int rowbase = blockIdx.x * 64;
  const int colbase = blockIdx.y * 128;
  const int rw = (tid >> 5) >> 2, cw = (tid >> 5) & 3;

  __builtin_prefetch(Wt + (size_t)colbase * DDIM, 0, 0);

  v8f acc[2][2] = {};
  gemm64x128(A + (size_t)rowbase * DDIM, Wt + (size_t)colbase * DDIM,
             DDIM, DDIM, DDIM, As, Bs, acc);

  const int lane = tid & 31;
  const int n    = lane & 15;
  const int mo   = (lane >> 4) * 8;
#pragma unroll
  for (int ci = 0; ci < 2; ++ci) {
    const int col = colbase + cw * 32 + ci * 16 + n;
    const float bn = bias[col];
#pragma unroll
    for (int ri = 0; ri < 2; ++ri) {
#pragma unroll
      for (int r = 0; r < 8; ++r) {
        const int row = rowbase + rw * 32 + ri * 16 + mo + r;
        OUT[(size_t)row * DDIM + col] = (__bf16)mishf(acc[ri][ci][r] + bn);
      }
    }
  }
}

// ---- row squared norms of bf16 [R,512]; 8 rows / 256-thread block --------

__global__ __launch_bounds__(256) void row_norms(const __bf16* __restrict__ E,
                                                 float* __restrict__ out) {
  const int row  = blockIdx.x * 8 + (threadIdx.x >> 5);
  const int lane = threadIdx.x & 31;
  const __bf16* p = E + (size_t)row * DDIM + lane * 16;
  v8bf a = *(const v8bf*)(p);
  v8bf b = *(const v8bf*)(p + 8);
  float s = 0.f;
#pragma unroll
  for (int i = 0; i < 8; ++i) {
    float u = (float)a[i], w = (float)b[i];
    s += u * u + w * w;
  }
#pragma unroll
  for (int o = 16; o; o >>= 1) s += __shfl_xor(s, o, 32);
  if (lane == 0) out[row] = s;
}

// ---- pairwise distance: DIST = sqrt(max(nx+ns-2*XE.SE^T,0)) as bf16 ------
// grid = (N/64, M/128), block = 256

__global__ __launch_bounds__(256)
void dist_kernel(const __bf16* __restrict__ XE, const __bf16* __restrict__ SE,
                 const float* __restrict__ nx, const float* __restrict__ ns,
                 __bf16* __restrict__ DIST) {
  __shared__ __align__(16) __bf16 As[2 * 64 * KC];
  __shared__ __align__(16) __bf16 Bs[2 * 128 * KC];
  const int tid     = threadIdx.x;
  const int rowbase = blockIdx.x * 64;
  const int colbase = blockIdx.y * 128;
  const int rw = (tid >> 5) >> 2, cw = (tid >> 5) & 3;

  __builtin_prefetch(SE + (size_t)colbase * DDIM, 0, 0);

  v8f acc[2][2] = {};
  gemm64x128(XE + (size_t)rowbase * DDIM, SE + (size_t)colbase * DDIM,
             DDIM, DDIM, DDIM, As, Bs, acc);

  const int lane = tid & 31;
  const int n    = lane & 15;
  const int mo   = (lane >> 4) * 8;
#pragma unroll
  for (int ci = 0; ci < 2; ++ci) {
    const int col = colbase + cw * 32 + ci * 16 + n;
    const float nsc = ns[col];
#pragma unroll
    for (int ri = 0; ri < 2; ++ri) {
#pragma unroll
      for (int r = 0; r < 8; ++r) {
        const int m = rowbase + rw * 32 + ri * 16 + mo + r;
        float d2 = nx[m] + nsc - 2.0f * acc[ri][ci][r];
        DIST[(size_t)m * M_ROWS + col] = (__bf16)sqrtf(fmaxf(d2, 0.0f));
      }
    }
  }
}

// ---- head: out = log_softmax(tanh(DIST @ Wp + bp)) -----------------------
// grid = N/16, block = 256 (8 waves); wave w owns padded cols [16w,16w+16).

__global__ __launch_bounds__(256)
void head_kernel(const __bf16* __restrict__ DIST, const __bf16* __restrict__ Wpt,
                 const float* __restrict__ bp, float* __restrict__ OUT) {
  __shared__ __align__(16) __bf16 Ds[16 * 256];
  __shared__ float zbuf[16 * CPAD];
  const int tid     = threadIdx.x;
  const int wave    = tid >> 5;
  const int lane    = tid & 31;
  const int rowbase = blockIdx.x * 16;
  const int colbase = wave * 16;

  v8f acc = {};
  for (int mb = 0; mb < M_ROWS; mb += 256) {
    float4* d = (float4*)Ds;
#pragma unroll
    for (int i = tid; i < 512; i += 256) {
      int rr = i >> 5;          // 32 float4 per 256-elem bf16 row
      int cc = i & 31;
      d[i] = *(const float4*)(DIST + (size_t)(rowbase + rr) * M_ROWS + mb + cc * 8);
    }
    __syncthreads();
    const __bf16* Bbase = Wpt + (size_t)colbase * M_ROWS + mb;
    __builtin_prefetch(Bbase, 0, 0);
#pragma unroll
    for (int kb = 0; kb < 256; kb += 32) {
      v16bf a = frag_ld(Ds, 256, kb);
      v16bf b = frag_ld(Bbase, M_ROWS, kb);
      acc = wmma_bf16(a, b, acc);
    }
    __syncthreads();
  }

  const int n   = lane & 15;
  const int mo  = (lane >> 4) * 8;
  const int col = colbase + n;
  const float bv = (col < CDIM) ? bp[col] : 0.0f;
#pragma unroll
  for (int r = 0; r < 8; ++r)
    zbuf[(mo + r) * CPAD + col] = tanhf(acc[r] + bv);
  __syncthreads();

#pragma unroll
  for (int rr = 0; rr < 2; ++rr) {
    const int row = wave * 2 + rr;
    float v[4];
    float mx = -3.4e38f;
#pragma unroll
    for (int t = 0; t < 4; ++t) {
      int c = lane + 32 * t;
      v[t] = (c < CDIM) ? zbuf[row * CPAD + c] : -3.4e38f;
      mx = fmaxf(mx, v[t]);
    }
#pragma unroll
    for (int o = 16; o; o >>= 1) mx = fmaxf(mx, __shfl_xor(mx, o, 32));
    float s = 0.f;
#pragma unroll
    for (int t = 0; t < 4; ++t) {
      int c = lane + 32 * t;
      if (c < CDIM) s += __expf(v[t] - mx);
    }
#pragma unroll
    for (int o = 16; o; o >>= 1) s += __shfl_xor(s, o, 32);
    const float lse = mx + __logf(s);
#pragma unroll
    for (int t = 0; t < 4; ++t) {
      int c = lane + 32 * t;
      if (c < CDIM)
        OUT[(size_t)(rowbase + row) * CDIM + c] = v[t] - lse;
    }
  }
}

// ---- host orchestration ---------------------------------------------------

extern "C" void kernel_launch(void* const* d_in, const int* in_sizes, int n_in,
                              void* d_out, int out_size, void* d_ws, size_t ws_size,
                              hipStream_t stream) {
  const float* x  = (const float*)d_in[0];
  const float* sm = (const float*)d_in[1];
  const float* W1 = (const float*)d_in[2];
  const float* b1 = (const float*)d_in[3];
  const float* W2 = (const float*)d_in[4];
  const float* b2 = (const float*)d_in[5];
  const float* Wp = (const float*)d_in[6];
  const float* bp = (const float*)d_in[7];
  float* out = (float*)d_out;

  // workspace carve-up (~90 MB)
  char* w = (char*)d_ws;
  auto carve = [&](size_t bytes) -> void* {
    void* p = (void*)w;
    w += (bytes + 255) & ~(size_t)255;
    return p;
  };
  __bf16* xA   = (__bf16*)carve((size_t)N_ROWS * DDIM * 2);   // x_bf -> xe
  __bf16* xB   = (__bf16*)carve((size_t)N_ROWS * DDIM * 2);   // h(x)
  __bf16* sA   = (__bf16*)carve((size_t)M_ROWS * DDIM * 2);   // s_bf -> se
  __bf16* sB   = (__bf16*)carve((size_t)M_ROWS * DDIM * 2);   // h(s)
  __bf16* W1t  = (__bf16*)carve((size_t)DDIM * DDIM * 2);
  __bf16* W2t  = (__bf16*)carve((size_t)DDIM * DDIM * 2);
  __bf16* Wpt  = (__bf16*)carve((size_t)CPAD * M_ROWS * 2);
  float*  nx   = (float*)carve((size_t)N_ROWS * 4);
  float*  ns   = (float*)carve((size_t)M_ROWS * 4);
  __bf16* DIST = (__bf16*)carve((size_t)N_ROWS * M_ROWS * 2);

  // 1) convert inputs to bf16
  cvt_bf16<<<(N_ROWS * DDIM) / 1024, 256, 0, stream>>>(x, xA, N_ROWS * DDIM);
  cvt_bf16<<<(M_ROWS * DDIM) / 1024, 256, 0, stream>>>(sm, sA, M_ROWS * DDIM);
  // 2) transposed bf16 weights
  transpose_cvt<<<(DDIM * DDIM) / 256, 256, 0, stream>>>(W1, W1t, DDIM, DDIM, DDIM);
  transpose_cvt<<<(DDIM * DDIM) / 256, 256, 0, stream>>>(W2, W2t, DDIM, DDIM, DDIM);
  transpose_cvt<<<(CPAD * M_ROWS) / 256, 256, 0, stream>>>(Wp, Wpt, M_ROWS, CDIM, CPAD);

  // 3) encoder: layer1 (A->B), layer2 (B->A)
  gemm_bias_mish<<<dim3(N_ROWS / 64, DDIM / 128), 256, 0, stream>>>(xA, W1t, b1, xB);
  gemm_bias_mish<<<dim3(M_ROWS / 64, DDIM / 128), 256, 0, stream>>>(sA, W1t, b1, sB);
  gemm_bias_mish<<<dim3(N_ROWS / 64, DDIM / 128), 256, 0, stream>>>(xB, W2t, b2, xA);
  gemm_bias_mish<<<dim3(M_ROWS / 64, DDIM / 128), 256, 0, stream>>>(sB, W2t, b2, sA);

  // 4) row norms
  row_norms<<<N_ROWS / 8, 256, 0, stream>>>(xA, nx);
  row_norms<<<M_ROWS / 8, 256, 0, stream>>>(sA, ns);

  // 5) pairwise distances (main 34 GF GEMM)
  dist_kernel<<<dim3(N_ROWS / 64, M_ROWS / 128), 256, 0, stream>>>(xA, sA, nx, ns, DIST);

  // 6) head GEMM + tanh + log_softmax
  head_kernel<<<N_ROWS / 16, 256, 0, stream>>>(DIST, Wpt, bp, out);
}
